// MambaBlock_64115271795231
// MI455X (gfx1250) — compile-verified, matrix operand order checked by reading, not run on hardware
//
#include <hip/hip_runtime.h>
#include <cstdint>
#include <cstddef>

// ---- problem constants (from reference) ----
#define D_MODEL   128
#define D_STATE   64
#define D_INNER   256
#define NHEADS    8
#define HEADDIM   32
#define CONV_DIM  384
#define D_IN_PROJ 648
#define W1_PAD    704            // 44 tiles of 16, zero padded
#define BATCH     8
#define SEQ       4096
#define ROWS      (BATCH * SEQ)  // 32768
#define EPS       1e-5f

// LDS panel pitches (elements): row + 8 ushort (4 dword) pad -> bank-group spread
#define P1        136            // for K=128 rows
#define P2        264            // for K=256 rows

typedef __attribute__((ext_vector_type(16))) __bf16 v16bf;
typedef __attribute__((ext_vector_type(8)))  float  v8f;
typedef __attribute__((ext_vector_type(4)))  unsigned int tdm_v4u;
typedef __attribute__((ext_vector_type(8)))  int          tdm_v8i;
typedef __attribute__((ext_vector_type(4)))  int          tdm_v4i;

#if defined(__has_builtin)
#if __has_builtin(__builtin_amdgcn_tensor_load_to_lds)
#define USE_TDM 1
#endif
#endif

union Frag16 { v16bf v; unsigned int u[8]; };

__device__ __forceinline__ unsigned short f32_to_bf16(float f) {
  unsigned int u = __float_as_uint(f);
  u += 0x7fffu + ((u >> 16) & 1u);          // round-to-nearest-even
  return (unsigned short)(u >> 16);
}

__device__ __forceinline__ float siluf(float v) { return v / (1.0f + expf(-v)); }

__device__ __forceinline__ float bcast_lane(float v, int lane) {
  return __uint_as_float(__builtin_amdgcn_readlane(__float_as_uint(v), lane));
}

// CDNA5 16-bit A/B fragment: VGPR j holds K pair base:
//   k = (j>=4 ? 16 : 0) + lanehalf*8 + (j&3)*2   (two consecutive k's per dword)
// j=0..3 and j=4..7 are each 16B-contiguous -> b128 loads (global or ds).
__device__ __forceinline__ int kpair(int j, int h) {
  return ((j & 4) << 2) + h * 8 + (j & 3) * 2;
}

__device__ __forceinline__ void load_frag(Frag16& f, const unsigned short* base, int h) {
#pragma unroll
  for (int j = 0; j < 8; ++j) f.u[j] = *(const unsigned int*)(base + kpair(j, h));
}

#ifdef USE_TDM
// Issue one 2D TDM load: 64 rows x ROWLEN bf16, global row stride = ROWLEN,
// LDS padded by 4 dwords per row (pad_interval covers exactly one row).
// PAD_IV: 5 -> every 64 dwords (ROWLEN=128), 6 -> every 128 dwords (ROWLEN=256).
template <int ROWLEN, int PAD_IV>
__device__ __forceinline__ void tdm_panel(const unsigned short* g, unsigned ldsOff) {
  unsigned long long ga = (unsigned long long)(size_t)g;
  tdm_v4u g0 = { 1u,                                   // count=1, user mode
                 ldsOff,                               // lds_addr (bytes)
                 (unsigned)ga,                         // global_addr[31:0]
                 ((unsigned)(ga >> 32) & 0x01FFFFFFu) | 0x80000000u }; // [56:32] | type=2
  tdm_v8i g1 = { (int)(0x10000u | (1u << 20) | ((unsigned)PAD_IV << 22) | (3u << 25)),
                 //    data_size=2B | pad_en  | pad_interval          | pad_amount=4dw
                 (int)((unsigned)ROWLEN << 16),        // tensor_dim0[15:0]
                 (int)(64u << 16),                     // tensor_dim1[15:0]
                 (int)((unsigned)ROWLEN << 16),        // tile_dim0
                 64,                                   // tile_dim1
                 ROWLEN,                               // tensor_dim0_stride[31:0]
                 0, 0 };
  tdm_v4i z4 = { 0, 0, 0, 0 };
#if __clang_major__ >= 23
  tdm_v8i z8 = { 0, 0, 0, 0, 0, 0, 0, 0 };
  __builtin_amdgcn_tensor_load_to_lds(g0, g1, z4, z4, z8, 0);
#else
  __builtin_amdgcn_tensor_load_to_lds(g0, g1, z4, z4, 0);
#endif
  __builtin_amdgcn_s_wait_tensorcnt(0);
}
#endif

// ---------------- weight conversion f32 -> bf16 (plain) ----------------
__global__ void k_f32_to_bf16(const float* __restrict__ src,
                              unsigned short* __restrict__ dst, int n) {
  int i = blockIdx.x * blockDim.x + threadIdx.x;
  if (i < n) dst[i] = f32_to_bf16(src[i]);
}

// ---------------- in_proj weight: [648,128] f32 -> [704,128] bf16, zero pad ----------------
__global__ void k_w1_pad_bf16(const float* __restrict__ src,
                              unsigned short* __restrict__ dst) {
  int i = blockIdx.x * blockDim.x + threadIdx.x;
  if (i >= W1_PAD * D_MODEL) return;
  int row = i / D_MODEL;
  dst[i] = (row < D_IN_PROJ) ? f32_to_bf16(src[i]) : (unsigned short)0;
}

// ---------------- RMSNorm(x) -> bf16 [ROWS,128] ----------------
__global__ void __launch_bounds__(256)
k_rmsnorm_bf16(const float* __restrict__ x, const float* __restrict__ w,
               unsigned short* __restrict__ xn) {
  int row  = blockIdx.x * 8 + (threadIdx.x >> 5);
  int lane = threadIdx.x & 31;
  const float* xr = x + (size_t)row * D_MODEL;
  float4 v = *(const float4*)(xr + lane * 4);
  float ss = v.x * v.x + v.y * v.y + v.z * v.z + v.w * v.w;
#pragma unroll
  for (int off = 16; off > 0; off >>= 1) ss += __shfl_xor(ss, off, 32);
  float sc = rsqrtf(ss * (1.0f / D_MODEL) + EPS);
  int k0 = lane * 4;
  unsigned short o0 = f32_to_bf16(v.x * sc * w[k0 + 0]);
  unsigned short o1 = f32_to_bf16(v.y * sc * w[k0 + 1]);
  unsigned short o2 = f32_to_bf16(v.z * sc * w[k0 + 2]);
  unsigned short o3 = f32_to_bf16(v.w * sc * w[k0 + 3]);
  uint2 st;
  st.x = (unsigned int)o0 | ((unsigned int)o1 << 16);
  st.y = (unsigned int)o2 | ((unsigned int)o3 << 16);
  *(uint2*)(xn + (size_t)row * D_MODEL + k0) = st;
}

// ---- GEMM1: Yproj[ROWS,648] = Xn @ W1^T ----
// Block: 8 waves = 8 row tiles, all sharing one 64-row B panel staged in LDS via TDM.
__global__ void __launch_bounds__(256)
k_gemm_inproj(const unsigned short* __restrict__ A,    // [ROWS,128] bf16
              const unsigned short* __restrict__ W,    // [704,128]  bf16 (padded)
              float* __restrict__ C) {                 // [ROWS,648] f32
  __shared__ __align__(16) unsigned short sB[64 * P1]; // 17 KB
  const int NG = 11;                                   // 44 col tiles / 4 per block
  int g  = blockIdx.x % NG;
  int mt = (blockIdx.x / NG) * 8 + (threadIdx.x >> 5);
  int lane = threadIdx.x & 31;
  int h = lane >> 4, m = lane & 15;

  const unsigned short* gPanel = W + (size_t)(g * 64) * D_MODEL;
#ifdef USE_TDM
  if ((threadIdx.x >> 5) == 0)
    tdm_panel<D_MODEL, 5>(gPanel, (unsigned)(size_t)&sB[0]);
#else
  for (int c = threadIdx.x; c < 64 * (D_MODEL / 8); c += 256) {
    int r = c / (D_MODEL / 8), wo = c % (D_MODEL / 8);
    *(uint4*)&sB[r * P1 + wo * 8] = *(const uint4*)(gPanel + r * D_MODEL + wo * 8);
  }
#endif
  __syncthreads();

  // hoist all A fragments (one clause of 8 b128 loads)
  const unsigned short* aBase = A + (size_t)(mt * 16 + m) * D_MODEL;
  Frag16 a[4];
#pragma unroll
  for (int ks = 0; ks < 4; ++ks) load_frag(a[ks], aBase + ks * 32, h);

  v8f acc[4] = {};
#pragma unroll
  for (int ks = 0; ks < 4; ++ks) {
#pragma unroll
    for (int t = 0; t < 4; ++t) {
      Frag16 b;
      load_frag(b, &sB[(t * 16 + m) * P1 + ks * 32], h);   // ds_load_b128 x2
      acc[t] = __builtin_amdgcn_wmma_f32_16x16x32_bf16(false, a[ks].v, false, b.v,
                                                       (short)0, acc[t], false, false);
    }
  }

#pragma unroll
  for (int t = 0; t < 4; ++t) {
    int n = g * 64 + t * 16 + m;
    if (n < D_IN_PROJ) {
#pragma unroll
      for (int r = 0; r < 8; ++r)
        C[(size_t)(mt * 16 + r + 8 * h) * D_IN_PROJ + n] = acc[t][r];
    }
  }
}

// ---------------- causal depthwise conv (K=4) + SiLU on xBC ----------------
__global__ void __launch_bounds__(256)
k_conv_silu(const float* __restrict__ Y, const float* __restrict__ cw,
            const float* __restrict__ cb, float* __restrict__ out) {
  int idx = blockIdx.x * blockDim.x + threadIdx.x;
  if (idx >= ROWS * CONV_DIM) return;
  int c = idx % CONV_DIM;
  int l = (idx / CONV_DIM) & (SEQ - 1);
  int b = idx / (CONV_DIM * SEQ);
  float acc = cb[c];
#pragma unroll
  for (int j = 0; j < 4; ++j) {
    int ls = l - 3 + j;
    float xv = (ls >= 0) ? Y[(size_t)(b * SEQ + ls) * D_IN_PROJ + D_INNER + c] : 0.0f;
    acc = fmaf(cw[c * 4 + j], xv, acc);
  }
  out[(size_t)(b * SEQ + l) * CONV_DIM + c] = siluf(acc);
}

// ---------------- SSM scan: one wave per (batch, head); lane = p ----------------
__global__ void __launch_bounds__(32)
k_ssm(const float* __restrict__ Y,      // dt raw at cols 640..647
      const float* __restrict__ cv,     // conv output [ROWS,384]
      const float* __restrict__ dt_bias, const float* __restrict__ A_log,
      const float* __restrict__ Dp, float* __restrict__ yout) {   // [ROWS,256]
  int hh = blockIdx.x & 7;
  int b  = blockIdx.x >> 3;
  int lane = threadIdx.x;               // p = lane (HEADDIM == 32)
  float Ah = -expf(A_log[hh]);
  float db = dt_bias[hh];
  float Dh = Dp[hh];
  float hst[64];                        // state h[p=lane, n=0..63] in VGPRs
#pragma unroll
  for (int n = 0; n < 64; ++n) hst[n] = 0.0f;

  for (int l = 0; l < SEQ; ++l) {
    size_t row = (size_t)b * SEQ + l;
    const float* cr = cv + row * CONV_DIM;
    if (l + 1 < SEQ) __builtin_prefetch(cr + CONV_DIM, 0, 1);   // next timestep
    float xv  = cr[hh * HEADDIM + lane];
    float blo = cr[D_INNER + lane];
    float bhi = cr[D_INNER + 32 + lane];
    float clo = cr[D_INNER + 64 + lane];
    float chi = cr[D_INNER + 96 + lane];
    float dtr = Y[row * D_IN_PROJ + 640 + hh] + db;
    float dt  = (dtr > 20.0f) ? dtr : log1pf(expf(dtr));   // softplus
    float dA  = expf(dt * Ah);
    float dtx = dt * xv;
    float y = 0.0f;
#pragma unroll
    for (int n = 0; n < 32; ++n) {
      float bn = bcast_lane(blo, n);    // v_readlane_b32 -> SGPR broadcast
      float cn = bcast_lane(clo, n);
      float hv = fmaf(hst[n], dA, dtx * bn);
      hst[n] = hv;
      y = fmaf(hv, cn, y);
    }
#pragma unroll
    for (int n = 0; n < 32; ++n) {
      float bn = bcast_lane(bhi, n);
      float cn = bcast_lane(chi, n);
      float hv = fmaf(hst[32 + n], dA, dtx * bn);
      hst[32 + n] = hv;
      y = fmaf(hv, cn, y);
    }
    yout[row * D_INNER + hh * HEADDIM + lane] = fmaf(Dh, xv, y);
  }
}

// ---------------- gate: rmsnorm(y * silu(z)) -> bf16 [ROWS,256] ----------------
__global__ void __launch_bounds__(256)
k_gate_norm(const float* __restrict__ Y, const float* __restrict__ yssm,
            const float* __restrict__ gw, unsigned short* __restrict__ G) {
  int row  = blockIdx.x * 8 + (threadIdx.x >> 5);
  int lane = threadIdx.x & 31;
  const float* zr = Y    + (size_t)row * D_IN_PROJ;   // z at cols 0..255
  const float* yr = yssm + (size_t)row * D_INNER;
  int k0 = lane * 8;
  float g[8];
  float ss = 0.0f;
#pragma unroll
  for (int i = 0; i < 8; ++i) {
    float z  = zr[k0 + i];
    float gv = yr[k0 + i] * siluf(z);
    g[i] = gv;
    ss = fmaf(gv, gv, ss);
  }
#pragma unroll
  for (int off = 16; off > 0; off >>= 1) ss += __shfl_xor(ss, off, 32);
  float sc = rsqrtf(ss * (1.0f / D_INNER) + EPS);
  uint4 st;
  unsigned int pk[4];
#pragma unroll
  for (int i = 0; i < 4; ++i) {
    unsigned short lo = f32_to_bf16(g[2 * i]     * sc * gw[k0 + 2 * i]);
    unsigned short hi = f32_to_bf16(g[2 * i + 1] * sc * gw[k0 + 2 * i + 1]);
    pk[i] = (unsigned int)lo | ((unsigned int)hi << 16);
  }
  st.x = pk[0]; st.y = pk[1]; st.z = pk[2]; st.w = pk[3];
  *(uint4*)(G + (size_t)row * D_INNER + k0) = st;
}

// ---- GEMM2 + residual: out = x + G @ W2^T; B panel (64x256) staged in LDS via TDM ----
__global__ void __launch_bounds__(256)
k_gemm_outproj(const unsigned short* __restrict__ A,   // G [ROWS,256] bf16
               const unsigned short* __restrict__ W,   // [128,256] bf16
               const float* __restrict__ x, float* __restrict__ out) {
  __shared__ __align__(16) unsigned short sB[64 * P2]; // 33 KB
  int g  = blockIdx.x & 1;                             // N = 128 -> 2 groups of 64
  int mt = (blockIdx.x >> 1) * 8 + (threadIdx.x >> 5);
  int lane = threadIdx.x & 31;
  int h = lane >> 4, m = lane & 15;

  const unsigned short* gPanel = W + (size_t)(g * 64) * D_INNER;
#ifdef USE_TDM
  if ((threadIdx.x >> 5) == 0)
    tdm_panel<D_INNER, 6>(gPanel, (unsigned)(size_t)&sB[0]);
#else
  for (int c = threadIdx.x; c < 64 * (D_INNER / 8); c += 256) {
    int r = c / (D_INNER / 8), wo = c % (D_INNER / 8);
    *(uint4*)&sB[r * P2 + wo * 8] = *(const uint4*)(gPanel + r * D_INNER + wo * 8);
  }
#endif
  __syncthreads();

  const unsigned short* aBase = A + (size_t)(mt * 16 + m) * D_INNER;
  Frag16 a[8];
#pragma unroll
  for (int ks = 0; ks < 8; ++ks) load_frag(a[ks], aBase + ks * 32, h);

  v8f acc[4] = {};
#pragma unroll
  for (int ks = 0; ks < 8; ++ks) {
#pragma unroll
    for (int t = 0; t < 4; ++t) {
      Frag16 b;
      load_frag(b, &sB[(t * 16 + m) * P2 + ks * 32], h);   // ds_load_b128 x2
      acc[t] = __builtin_amdgcn_wmma_f32_16x16x32_bf16(false, a[ks].v, false, b.v,
                                                       (short)0, acc[t], false, false);
    }
  }

#pragma unroll
  for (int t = 0; t < 4; ++t) {
    int n = g * 64 + t * 16 + m;
#pragma unroll
    for (int r = 0; r < 8; ++r) {
      size_t row = (size_t)(mt * 16 + r + 8 * h);
      out[row * D_MODEL + n] = x[row * D_MODEL + n] + acc[t][r];
    }
  }
}

extern "C" void kernel_launch(void* const* d_in, const int* in_sizes, int n_in,
                              void* d_out, int out_size, void* d_ws, size_t ws_size,
                              hipStream_t stream) {
  const float* x        = (const float*)d_in[0];
  const float* norm_w   = (const float*)d_in[1];
  const float* in_proj  = (const float*)d_in[2];
  const float* conv_w   = (const float*)d_in[3];
  const float* conv_b   = (const float*)d_in[4];
  const float* dt_bias  = (const float*)d_in[5];
  const float* A_log    = (const float*)d_in[6];
  const float* Dp       = (const float*)d_in[7];
  const float* gnorm_w  = (const float*)d_in[8];
  const float* out_proj = (const float*)d_in[9];
  float* out = (float*)d_out;
  (void)in_sizes; (void)n_in; (void)out_size; (void)ws_size;

  // workspace carve-out (256B aligned), ~195 MB total
  char* ws = (char*)d_ws;
  size_t off = 0;
  auto carve = [&](size_t bytes) -> void* {
    void* p = ws + off;
    off += (bytes + 255) & ~(size_t)255;
    return p;
  };
  unsigned short* Xn   = (unsigned short*)carve((size_t)ROWS * D_MODEL * 2);
  unsigned short* W1   = (unsigned short*)carve((size_t)W1_PAD * D_MODEL * 2);
  unsigned short* W2   = (unsigned short*)carve((size_t)D_MODEL * D_INNER * 2);
  float*          Yp   = (float*)carve((size_t)ROWS * D_IN_PROJ * 4);
  float*          Cv   = (float*)carve((size_t)ROWS * CONV_DIM * 4);
  float*          Ys   = (float*)carve((size_t)ROWS * D_INNER * 4);
  unsigned short* G    = (unsigned short*)carve((size_t)ROWS * D_INNER * 2);

  k_w1_pad_bf16<<<(W1_PAD * D_MODEL + 255) / 256, 256, 0, stream>>>(in_proj, W1);
  k_f32_to_bf16<<<(D_MODEL * D_INNER + 255) / 256, 256, 0, stream>>>(out_proj, W2, D_MODEL * D_INNER);
  k_rmsnorm_bf16<<<ROWS / 8, 256, 0, stream>>>(x, norm_w, Xn);
  k_gemm_inproj<<<(ROWS / 128) * 11, 256, 0, stream>>>(Xn, W1, Yp);
  k_conv_silu<<<(ROWS * CONV_DIM) / 256, 256, 0, stream>>>(Yp, conv_w, conv_b, Cv);
  k_ssm<<<BATCH * NHEADS, 32, 0, stream>>>(Yp, Cv, dt_bias, A_log, Dp, Ys);
  k_gate_norm<<<ROWS / 8, 256, 0, stream>>>(Yp, Ys, gnorm_w, G);
  k_gemm_outproj<<<(ROWS / 128) * 2, 256, 0, stream>>>(G, W2, x, out);
}
